// SlicedWassersteinPooling_76708115906638
// MI455X (gfx1250) — matrix operand order repaired and checked by reading, not compiled, not run on hardware
//
#include <hip/hip_runtime.h>
#include <hip/hip_bf16.h>

// ---------------------------------------------------------------------------
// Sliced-Wasserstein pooling for MI455X (gfx1250, wave32, WMMA).
// Roofline: ~550 MiB DRAM traffic => ~24us floor @23.3TB/s. GEMM = 12.9 GFLOP
// done in bf16 WMMA (f32 accum) => compute stays far below memory time.
// ---------------------------------------------------------------------------

#define BB 16
#define SS 4096
#define HH 768
#define PP 128
#define BM 128   // GEMM block tile rows
#define BK 32    // GEMM K step (matches v_wmma_f32_16x16x32_bf16)

typedef __bf16        bf16x16 __attribute__((ext_vector_type(16)));
typedef float         f32x8   __attribute__((ext_vector_type(8)));
typedef float         f32x4   __attribute__((ext_vector_type(4)));
typedef unsigned int  u32x4   __attribute__((ext_vector_type(4)));

union FragAB { u32x4 u[2]; bf16x16 v; };

__device__ __forceinline__ unsigned short f2bf(float f) {
  unsigned int u = __float_as_uint(f);
  unsigned int r = u + 0x7fffu + ((u >> 16) & 1u);   // round-to-nearest-even
  return (unsigned short)(r >> 16);
}

// ---------------------------------------------------------------------------
// Kernel 1: row-normalize projections, emit bf16 [P][H]
// ---------------------------------------------------------------------------
__global__ __launch_bounds__(256) void k_norm_proj(const float* __restrict__ proj,
                                                   unsigned short* __restrict__ projb) {
  const int p = blockIdx.x, t = threadIdx.x;
  __shared__ float red[256];
  float s = 0.f;
  for (int i = t; i < HH; i += 256) { float x = proj[p * HH + i]; s += x * x; }
  red[t] = s; __syncthreads();
  for (int o = 128; o > 0; o >>= 1) { if (t < o) red[t] += red[t + o]; __syncthreads(); }
  const float inv = rsqrtf(red[0]);
  for (int i = t; i < HH; i += 256)
    projb[p * HH + i] = f2bf(proj[p * HH + i] * inv);
}

// ---------------------------------------------------------------------------
// Kernel 2: projected[(b*S+s)][p] = sum_h emb*mask * proj_n  via bf16 WMMA.
// Block: 128 rows x 128 cols, 8 waves (wave w owns M-tile w, all 8 N-tiles).
// A tile: f32 global -> mask+convert in VALU -> LDS (DScnt path).
// B tile: bf16 global -> LDS directly via global_load_async_to_lds_b128
//         (ASYNCcnt path; data is L2-resident so this is pure plumbing win).
// LDS rows use a 40-ushort (80B) stride so every b128 access is 16B aligned.
// Fragment layouts per CDNA5 ISA:
//   A 16x32 bf16 : lanes<16 -> K {0..7,16..23}; lanes>=16 -> K {8..15,24..31}
//   B 32x16 bf16 : lanes<16 -> K {0..7,16..23}; lanes>=16 -> K {8..15,24..31}
// All 8 B fragments are preloaded so the 8 WMMAs issue back-to-back behind a
// single s_wait_dscnt instead of one wait per WMMA.
// ---------------------------------------------------------------------------
__global__ __launch_bounds__(256) void k_gemm(const float* __restrict__ emb,
                                              const int* __restrict__ mask,
                                              const unsigned short* __restrict__ projb,
                                              float* __restrict__ projected) {
  __shared__ __align__(16) unsigned short As[BM][40];
  __shared__ __align__(16) unsigned short Bs[PP][40];

  const int t = threadIdx.x;
  const int wave = t >> 5, lane = t & 31;
  const int hi = lane >> 4, l16 = lane & 15;
  const long row0 = (long)blockIdx.x * BM;

  const f32x8 zero = {0.f, 0.f, 0.f, 0.f, 0.f, 0.f, 0.f, 0.f};
  f32x8 acc[8];
#pragma unroll
  for (int n = 0; n < 8; ++n) acc[n] = zero;

  for (int kk = 0; kk < HH; kk += BK) {
    // ---- stage B: async DMA bf16 slice straight into LDS (no VGPR bounce)
#pragma unroll
    for (int it = 0; it < 2; ++it) {
      const int idx = t + it * 256;           // 512 chunks of 16 ushorts
      const int r = idx >> 1, c8 = (idx & 1) << 4;
      const unsigned short* gsrc = projb + (long)r * HH + kk + c8;
      const unsigned ldst = (unsigned)(uintptr_t)&Bs[r][c8];
      asm volatile("global_load_async_to_lds_b128 %0, %1, off"
                   :: "v"(ldst), "v"(gsrc) : "memory");
    }
    // ---- stage A: 128x32 f32 -> bf16 (apply mask), coalesced float4 loads
#pragma unroll
    for (int it = 0; it < 4; ++it) {
      const int idx = t + it * 256;           // 1024 float4 slots
      const int r = idx >> 3, c4 = (idx & 7) << 2;
      const long grow = row0 + r;
      const float mf = (float)mask[grow];
      const f32x4 x = *(const f32x4*)(emb + grow * HH + kk + c4);
      As[r][c4 + 0] = f2bf(x.x * mf);
      As[r][c4 + 1] = f2bf(x.y * mf);
      As[r][c4 + 2] = f2bf(x.z * mf);
      As[r][c4 + 3] = f2bf(x.w * mf);
    }
    asm volatile("s_wait_asynccnt 0x0" ::: "memory");
    __syncthreads();

    FragAB a;
    a.u[0] = *(const u32x4*)&As[wave * 16 + l16][hi * 8];
    a.u[1] = *(const u32x4*)&As[wave * 16 + l16][16 + hi * 8];
    FragAB bf[8];
#pragma unroll
    for (int n = 0; n < 8; ++n) {
      bf[n].u[0] = *(const u32x4*)&Bs[n * 16 + l16][hi * 8];
      bf[n].u[1] = *(const u32x4*)&Bs[n * 16 + l16][16 + hi * 8];
    }
#pragma unroll
    for (int n = 0; n < 8; ++n)
      acc[n] = __builtin_amdgcn_wmma_f32_16x16x32_bf16(
          false, a.v, false, bf[n].v, (short)0, acc[n], false, false);
    __syncthreads();
  }

  // C/D layout: VGPR r -> row = r + 8*hi, col = l16 (per 16x16 tile)
#pragma unroll
  for (int n = 0; n < 8; ++n)
#pragma unroll
    for (int r = 0; r < 8; ++r) {
      const long row = row0 + wave * 16 + r + hi * 8;
      projected[row * PP + n * 16 + l16] = acc[n][r];
    }
}

// ---------------------------------------------------------------------------
// Kernel 3: per (b,p): bitonic-sort the S column in LDS, then Gaussian
// weighted quantile -> weighted_proj[b][p]
// ---------------------------------------------------------------------------
__global__ __launch_bounds__(512) void k_sort_quantile(const float* __restrict__ projected,
                                                       const int* __restrict__ mask,
                                                       float* __restrict__ wq) {
  const int p = blockIdx.x, b = blockIdx.y, t = threadIdx.x;
  __shared__ float v[SS];
  __shared__ float red[512];

  for (int i = t; i < SS; i += 512)
    v[i] = projected[((long)(b * SS + i)) * PP + p];

  float cnt = 0.f;
  for (int i = t; i < SS; i += 512) cnt += (float)mask[b * SS + i];
  red[t] = cnt; __syncthreads();
  for (int o = 256; o > 0; o >>= 1) { if (t < o) red[t] += red[t + o]; __syncthreads(); }
  const float validf = red[0];
  __syncthreads();

  // bitonic sort, ascending
  for (int k = 2; k <= SS; k <<= 1)
    for (int j = k >> 1; j > 0; j >>= 1) {
      __syncthreads();
      for (int i = t; i < SS; i += 512) {
        const int ixj = i ^ j;
        if (ixj > i) {
          const float a = v[i], c = v[ixj];
          const bool up = ((i & k) == 0);
          if ((a > c) == up) { v[i] = c; v[ixj] = a; }
        }
      }
    }
  __syncthreads();

  const float center = validf * 0.5f;
  const float invden = 1.f / validf;          // TEMPERATURE = 1
  float wsum = 0.f, wdot = 0.f;
  for (int i = t; i < SS; i += 512) {
    const float d = (float)i - center;
    const float w = __expf(-d * d * invden) * (float)mask[b * SS + i];
    wsum += w; wdot += w * v[i];
  }
  red[t] = wsum; __syncthreads();
  for (int o = 256; o > 0; o >>= 1) { if (t < o) red[t] += red[t + o]; __syncthreads(); }
  wsum = red[0]; __syncthreads();
  red[t] = wdot; __syncthreads();
  for (int o = 256; o > 0; o >>= 1) { if (t < o) red[t] += red[t + o]; __syncthreads(); }
  if (t == 0) wq[b * PP + p] = red[0] / (wsum + 1e-9f);
}

// ---------------------------------------------------------------------------
// Kernel 4: scores = -||projected - wq||, masked softmax over S -> attn
// One block per batch; one wave32 per row s (coalesced 512B row reads).
// ---------------------------------------------------------------------------
__global__ __launch_bounds__(256) void k_scores_softmax(const float* __restrict__ projected,
                                                        const float* __restrict__ wq,
                                                        const int* __restrict__ mask,
                                                        float* __restrict__ attn) {
  const int b = blockIdx.x, t = threadIdx.x;
  const int wave = t >> 5, lane = t & 31;
  __shared__ float wred[8];
  __shared__ float red[256];

  const f32x4 q = *(const f32x4*)(wq + b * PP + lane * 4);
  float mmax = -INFINITY;
  for (int s = wave; s < SS; s += 8) {
    const f32x4 pr = *(const f32x4*)(projected + ((long)(b * SS + s)) * PP + lane * 4);
    const f32x4 d = pr - q;
    float ss = d.x * d.x + d.y * d.y + d.z * d.z + d.w * d.w;
#pragma unroll
    for (int o = 16; o > 0; o >>= 1) ss += __shfl_xor(ss, o, 32);
    const float sc = mask[b * SS + s] ? -sqrtf(ss) : -INFINITY;
    if (lane == 0) attn[b * SS + s] = sc;
    mmax = fmaxf(mmax, sc);
  }
  if (lane == 0) wred[wave] = mmax;
  __syncthreads();
  float gmax = -INFINITY;
#pragma unroll
  for (int i = 0; i < 8; ++i) gmax = fmaxf(gmax, wred[i]);

  float psum = 0.f;
  for (int s = t; s < SS; s += 256) {
    const float e = __expf(attn[b * SS + s] - gmax);
    attn[b * SS + s] = e;
    psum += e;
  }
  red[t] = psum; __syncthreads();
  for (int o = 128; o > 0; o >>= 1) { if (t < o) red[t] += red[t + o]; __syncthreads(); }
  const float inv = 1.f / red[0];
  for (int s = t; s < SS; s += 256) attn[b * SS + s] *= inv;
}

// ---------------------------------------------------------------------------
// Kernel 5/6: zero output, then pooled[b][h] += sum_s attn*emb (atomic f32)
// ---------------------------------------------------------------------------
__global__ void k_zero(float* __restrict__ out, int n) {
  const int i = blockIdx.x * 256 + threadIdx.x;
  if (i < n) out[i] = 0.f;
}

__global__ __launch_bounds__(256) void k_pool(const float* __restrict__ attn,
                                              const float* __restrict__ emb,
                                              const int* __restrict__ mask,
                                              float* __restrict__ out) {
  const int b = blockIdx.x, hbase = blockIdx.y * 256, s0 = blockIdx.z * 512;
  __shared__ float a[512];
  for (int i = threadIdx.x; i < 512; i += 256)
    a[i] = attn[b * SS + s0 + i] * (float)mask[b * SS + s0 + i];
  __syncthreads();
  const int h = hbase + threadIdx.x;
  float acc = 0.f;
  for (int i = 0; i < 512; ++i)
    acc = fmaf(a[i], emb[((long)(b * SS + s0 + i)) * HH + h], acc);
  atomicAdd(&out[b * HH + h], acc);
}

// ---------------------------------------------------------------------------
extern "C" void kernel_launch(void* const* d_in, const int* in_sizes, int n_in,
                              void* d_out, int out_size, void* d_ws, size_t ws_size,
                              hipStream_t stream) {
  const float* emb  = (const float*)d_in[0];   // (B,S,H) f32
  const float* proj = (const float*)d_in[1];   // (P,H)   f32
  const int*   mask = (const int*)d_in[2];     // (B,S)   i32
  float* out = (float*)d_out;                  // (B,H)   f32

  // workspace layout (~34.1 MiB)
  char* ws = (char*)d_ws;
  unsigned short* projb  = (unsigned short*)ws;                         // P*H bf16
  float* projected = (float*)(ws + (256u << 10));                      // B*S*P f32
  float* wq        = (float*)(ws + (256u << 10) + (size_t)BB * SS * PP * 4);
  float* attn      = (float*)(ws + (256u << 10) + (size_t)BB * SS * PP * 4 + 8192);

  k_norm_proj<<<PP, 256, 0, stream>>>(proj, projb);
  k_gemm<<<(BB * SS) / BM, 256, 0, stream>>>(emb, mask, projb, projected);
  k_sort_quantile<<<dim3(PP, BB), 512, 0, stream>>>(projected, mask, wq);
  k_scores_softmax<<<BB, 256, 0, stream>>>(projected, wq, mask, attn);
  k_zero<<<(BB * HH + 255) / 256, 256, 0, stream>>>(out, BB * HH);
  k_pool<<<dim3(BB, HH / 256, SS / 512), 256, 0, stream>>>(attn, emb, mask, out);
}
